// PyTorchAttention_69715909149069
// MI455X (gfx1250) — matrix operand order repaired and checked by reading, not compiled, not run on hardware
//
#include <hip/hip_runtime.h>
#include <hip/hip_bf16.h>

// ---------------------------------------------------------------------------
// Causal MHA for MI455X (gfx1250), wave32, V_WMMA_F32_16X16X32_BF16 throughout.
// - All matmul B-operands pre-transposed: every WMMA fragment load is
//   contiguous (2 x global_load_b128 per fragment) -> WMMA-issue dominance.
// - Causal mask applied only in the single diagonal-straddling key block;
//   steady-state loop is mask-free.
// - Softmax scale (1/sqrt(Dh) * log2 e) folded into W_Q at conversion time;
//   softmax runs in the exp2 domain (v_exp_f32 is natively 2^x).
// ---------------------------------------------------------------------------

typedef __bf16 bf16_t;
typedef __attribute__((ext_vector_type(16))) __bf16 v16bf;
typedef __attribute__((ext_vector_type(4)))  __bf16 v4bf;
typedef __attribute__((ext_vector_type(4)))  float  v4f;
typedef __attribute__((ext_vector_type(8)))  float  v8f;

#define D_MODEL 1024
#define N_HEADS 16
#define D_HEAD  64
#define SEQ     4096

// --- WMMA fragment index maps (CDNA5 ISA 7.12.2) ---------------------------
// 16-bit A-matrix 16x32: lanes 0-15 hold M=0..15 ; VGPR v holds K pairs,
// code uses contiguous K: lanes 0-15 hold K=0..15, lanes 16-31 hold K=16..31 (same order as B).
__device__ __forceinline__ int a_kk(int v, int hl) {
  return 2 * v + 16 * hl;
}
// 16-bit B-matrix 32x16 (contiguous K order used here, not ISA table order): lanes 0-15 hold
// K=0..15 across the 8 VGPRs, lanes 16-31 hold K=16..31.
__device__ __forceinline__ int b_kk(int v, int hl) {
  return 2 * v + 16 * hl;
}

__device__ __forceinline__ v8f wmma_bf16(v16bf a, v16bf b, v8f c) {
  // (neg_a, A, neg_b, B, c_mod, C, reuse_a, reuse_b)
  return __builtin_amdgcn_wmma_f32_16x16x32_bf16(false, a, false, b,
                                                 (short)0, c, false, false);
}

// ---------------------------------------------------------------------------
// fp32 -> bf16 conversion, 4 elements / lane (global_load_b128 / store_b64)
// ---------------------------------------------------------------------------
__global__ __launch_bounds__(256) void cvt_f32_bf16(
    const float* __restrict__ src, bf16_t* __restrict__ dst, int n4) {
  int i = blockIdx.x * blockDim.x + threadIdx.x;
  if (i < n4) {
    v4f f = ((const v4f*)src)[i];
    v4bf b;
    b[0] = (bf16_t)f[0];
    b[1] = (bf16_t)f[1];
    b[2] = (bf16_t)f[2];
    b[3] = (bf16_t)f[3];
    ((v4bf*)dst)[i] = b;
  }
}

// ---------------------------------------------------------------------------
// fp32 -> bf16 scaled transpose: dst[c][r] = src[r][c] * scale; 32x32 LDS
// tile, both global sides coalesced. One-time cost per weight matrix.
// ---------------------------------------------------------------------------
__global__ __launch_bounds__(256) void cvt_t_f32_bf16(
    const float* __restrict__ src, bf16_t* __restrict__ dst, int rows,
    int cols, float scale) {
  __shared__ float t[32][33];
  const int c0 = blockIdx.x * 32, r0 = blockIdx.y * 32;
  const int tx = threadIdx.x, ty = threadIdx.y;  // (32, 8)
#pragma unroll
  for (int i = ty; i < 32; i += 8)
    t[i][tx] = src[(size_t)(r0 + i) * cols + c0 + tx] * scale;
  __syncthreads();
#pragma unroll
  for (int i = ty; i < 32; i += 8)
    dst[(size_t)(c0 + i) * rows + r0 + tx] = (bf16_t)t[tx][i];
}

// ---------------------------------------------------------------------------
// C[M,N] = A[M,K] @ Bt[N,K]^T ; A row-major bf16, Bt = B transposed (row-major
// [N][K]) so every B-fragment is 16 contiguous bf16.
// Block: 256 threads = 8 waves; block tile 128(M) x 64(N).
// Each wave: 16(M) x 64(N) strip -> 4 accumulator frags sharing one A frag.
// TRANSOUT: store C transposed (used to produce Vt[d][s] for attention).
// ---------------------------------------------------------------------------
template <bool BF16OUT, bool TRANSOUT>
__global__ __launch_bounds__(256) void gemm_bf16_wmma(
    const bf16_t* __restrict__ A, const bf16_t* __restrict__ Bt,
    bf16_t* __restrict__ Cb, float* __restrict__ Cf, int M, int N, int K) {
  const int lane = threadIdx.x & 31;
  const int wid  = threadIdx.x >> 5;
  const int hl   = lane >> 4;
  const int ln   = lane & 15;

  const int m0 = blockIdx.y * 128 + wid * 16;
  const int n0 = blockIdx.x * 64;

  v8f acc[4];
#pragma unroll
  for (int j = 0; j < 4; ++j)
#pragma unroll
    for (int r = 0; r < 8; ++r) acc[j][r] = 0.0f;

  for (int k0 = 0; k0 < K; k0 += 32) {
    v16bf a;
#pragma unroll
    for (int v = 0; v < 8; ++v) {
      const int kk = a_kk(v, hl);
      const bf16_t* p = A + (size_t)(m0 + ln) * K + k0 + kk;
      a[2 * v]     = p[0];
      a[2 * v + 1] = p[1];
    }
#pragma unroll
    for (int j = 0; j < 4; ++j) {
      v16bf b;
#pragma unroll
      for (int v = 0; v < 8; ++v) {
        const int kk = b_kk(v, hl);  // contiguous in Bt row
        const bf16_t* p = Bt + (size_t)(n0 + 16 * j + ln) * K + k0 + kk;
        b[2 * v]     = p[0];
        b[2 * v + 1] = p[1];
      }
      acc[j] = wmma_bf16(a, b, acc[j]);
    }
  }

  // C/D layout: lanes 0-15 -> M=r, N=lane ; lanes 16-31 -> M=8+r, N=lane-16
  const int mbase = m0 + 8 * hl;
#pragma unroll
  for (int j = 0; j < 4; ++j)
#pragma unroll
    for (int r = 0; r < 8; ++r) {
      const int row = mbase + r;
      const int col = n0 + 16 * j + ln;
      const size_t idx =
          TRANSOUT ? (size_t)col * M + row : (size_t)row * N + col;
      if (BF16OUT)
        Cb[idx] = (bf16_t)acc[j][r];
      else
        Cf[idx] = acc[j][r];
    }
}

// ---------------------------------------------------------------------------
// One 32-wide key block of flash attention for a 16-row query tile.
// MASKED=true only for the single block straddling the causal diagonal.
// ---------------------------------------------------------------------------
template <bool MASKED>
__device__ __forceinline__ void attn_block(
    int kb, int qb, int hl, int ln, int hoff, const bf16_t* __restrict__ K,
    const bf16_t* __restrict__ Vt, const v16bf (&qA)[2], v8f (&o)[4],
    float (&mrow)[8], float (&lrow)[8], bf16_t* __restrict__ Pw) {
  // ---- scores: S(16x32) = Q(16x64) @ K_blk^T(64x32), two 16x16 f32 tiles
  v8f s0, s1;
#pragma unroll
  for (int r = 0; r < 8; ++r) { s0[r] = 0.0f; s1[r] = 0.0f; }
#pragma unroll
  for (int s = 0; s < 2; ++s) {
    v16bf b0, b1;  // B[k][n] = K[kb + n (+16)][32*s + k] -> contiguous loads
#pragma unroll
    for (int v = 0; v < 8; ++v) {
      const int kk = b_kk(v, hl) + 32 * s;
      const bf16_t* p0 = K + (size_t)(kb + ln)      * D_MODEL + hoff + kk;
      const bf16_t* p1 = K + (size_t)(kb + 16 + ln) * D_MODEL + hoff + kk;
      b0[2 * v] = p0[0]; b0[2 * v + 1] = p0[1];
      b1[2 * v] = p1[0]; b1[2 * v + 1] = p1[1];
    }
    s0 = wmma_bf16(qA[s], b0, s0);
    s1 = wmma_bf16(qA[s], b1, s1);
  }

  // ---- online softmax update (row r lives in lanes of one 16-lane half)
  // Q was pre-scaled by 1/sqrt(Dh)*log2(e): scores are already exp2-domain.
  const int qrow0 = qb + 8 * hl;
#pragma unroll
  for (int r = 0; r < 8; ++r) {
    float x0 = s0[r];
    float x1 = s1[r];
    if (MASKED) {
      const int q = qrow0 + r;
      if (kb + ln > q)      x0 = -1e30f;
      if (kb + 16 + ln > q) x1 = -1e30f;
    }
    float mx = fmaxf(x0, x1);
#pragma unroll
    for (int off = 1; off < 16; off <<= 1)
      mx = fmaxf(mx, __shfl_xor(mx, off, 32));
    const float mnew  = fmaxf(mrow[r], mx);
    const float alpha = exp2f(mrow[r] - mnew);
    const float p0 = exp2f(x0 - mnew);
    const float p1 = exp2f(x1 - mnew);
    float ps = p0 + p1;
#pragma unroll
    for (int off = 1; off < 16; off <<= 1) ps += __shfl_xor(ps, off, 32);
    lrow[r] = lrow[r] * alpha + ps;
    mrow[r] = mnew;
#pragma unroll
    for (int j = 0; j < 4; ++j) o[j][r] *= alpha;
    const int prow = r + 8 * hl;
    Pw[prow * 32 + ln]      = (bf16_t)p0;
    Pw[prow * 32 + 16 + ln] = (bf16_t)p1;
  }
  // per-wave LDS region; DS ops are in-order per wave -- fence the compiler
  // and drain DScnt before re-reading in transposed layout.
  __builtin_amdgcn_wave_barrier();
  asm volatile("s_wait_dscnt 0" ::: "memory");

  // ---- P (16x32) as A-frag from LDS
  v16bf pA;
#pragma unroll
  for (int v = 0; v < 8; ++v) {
    const int kk = a_kk(v, hl);
    pA[2 * v]     = Pw[ln * 32 + kk];
    pA[2 * v + 1] = Pw[ln * 32 + kk + 1];
  }
  __builtin_amdgcn_wave_barrier();

  // ---- O(16x64) += P(16x32) @ V_blk(32x64); Vt[d][s] -> contiguous frags
#pragma unroll
  for (int j = 0; j < 4; ++j) {
    v16bf bv;
#pragma unroll
    for (int v = 0; v < 8; ++v) {
      const int kk = b_kk(v, hl);
      const bf16_t* p = Vt + (size_t)(hoff + 16 * j + ln) * SEQ + kb + kk;
      bv[2 * v]     = p[0];
      bv[2 * v + 1] = p[1];
    }
    o[j] = wmma_bf16(pA, bv, o[j]);
  }
}

// ---------------------------------------------------------------------------
// Flash-style causal attention. One wave per (head, 16-query tile).
// Q,K bf16 [SEQ, D_MODEL] (head h in cols h*64..h*64+63); V transposed as
// Vt [D_MODEL, SEQ]. Steady-state key blocks are unmasked; exactly one
// diagonal block per wave takes the masked path.
// ---------------------------------------------------------------------------
__global__ __launch_bounds__(256) void attn_causal_wmma(
    const bf16_t* __restrict__ Q, const bf16_t* __restrict__ K,
    const bf16_t* __restrict__ Vt, bf16_t* __restrict__ O) {
  __shared__ bf16_t Pst[8][16 * 32];  // per-wave P staging (C-layout -> A-layout)

  const int lane = threadIdx.x & 31;
  const int wid  = threadIdx.x >> 5;
  const int hl   = lane >> 4;
  const int ln   = lane & 15;

  const int gw    = blockIdx.x * 8 + wid;  // global wave id
  const int head  = gw >> 8;               // 256 query tiles per head
  const int qtile = gw & 255;
  const int qb    = qtile * 16;
  const int hoff  = head * D_HEAD;
  bf16_t* Pw = &Pst[wid][0];

  // Q tile (16 x 64) as two 16x32 A-frags, kept in registers for whole loop.
  v16bf qA[2];
#pragma unroll
  for (int s = 0; s < 2; ++s)
#pragma unroll
    for (int v = 0; v < 8; ++v) {
      const int kk = a_kk(v, hl) + 32 * s;
      const bf16_t* p = Q + (size_t)(qb + ln) * D_MODEL + hoff + kk;
      qA[s][2 * v]     = p[0];
      qA[s][2 * v + 1] = p[1];
    }

  v8f o[4];
  float mrow[8], lrow[8];
#pragma unroll
  for (int j = 0; j < 4; ++j)
#pragma unroll
    for (int r = 0; r < 8; ++r) o[j][r] = 0.0f;
#pragma unroll
  for (int r = 0; r < 8; ++r) {
    mrow[r] = -1e30f;
    lrow[r] = 0.0f;
  }

  // Block [kb, kb+31] is entirely below the diagonal iff kb+31 <= qb; the one
  // block at kb = (qb/32)*32 straddles it and is the only one needing a mask.
  const int kb_mask = (qb / 32) * 32;
  for (int kb = 0; kb < kb_mask; kb += 32)
    attn_block<false>(kb, qb, hl, ln, hoff, K, Vt, qA, o, mrow, lrow, Pw);
  attn_block<true>(kb_mask, qb, hl, ln, hoff, K, Vt, qA, o, mrow, lrow, Pw);

  // ---- normalize and store (bf16 into [SEQ, D_MODEL] at head columns)
#pragma unroll
  for (int j = 0; j < 4; ++j)
#pragma unroll
    for (int r = 0; r < 8; ++r) {
      const int row = qb + 8 * hl + r;
      O[(size_t)row * D_MODEL + hoff + 16 * j + ln] =
          (bf16_t)(o[j][r] / lrow[r]);
    }
}

// ---------------------------------------------------------------------------
extern "C" void kernel_launch(void* const* d_in, const int* in_sizes, int n_in,
                              void* d_out, int out_size, void* d_ws,
                              size_t ws_size, hipStream_t stream) {
  (void)in_sizes; (void)n_in; (void)out_size; (void)ws_size;

  const float* x  = (const float*)d_in[0];
  const float* Wq = (const float*)d_in[1];
  const float* Wk = (const float*)d_in[2];
  const float* Wv = (const float*)d_in[3];
  const float* Wo = (const float*)d_in[4];
  float* out = (float*)d_out;

  const size_t SZ_X = (size_t)SEQ * D_MODEL;      // 4M elems
  const size_t SZ_W = (size_t)D_MODEL * D_MODEL;  // 1M elems

  char* ws = (char*)d_ws;
  bf16_t* xb  = (bf16_t*)ws; ws += SZ_X * sizeof(bf16_t);
  bf16_t* wqT = (bf16_t*)ws; ws += SZ_W * sizeof(bf16_t);
  bf16_t* wkT = (bf16_t*)ws; ws += SZ_W * sizeof(bf16_t);
  bf16_t* wvT = (bf16_t*)ws; ws += SZ_W * sizeof(bf16_t);
  bf16_t* woT = (bf16_t*)ws; ws += SZ_W * sizeof(bf16_t);
  bf16_t* Qb  = (bf16_t*)ws; ws += SZ_X * sizeof(bf16_t);
  bf16_t* Kb  = (bf16_t*)ws; ws += SZ_X * sizeof(bf16_t);
  bf16_t* VbT = (bf16_t*)ws; ws += SZ_X * sizeof(bf16_t);  // Vt [D_MODEL][SEQ]
  bf16_t* Ab  = (bf16_t*)ws; ws += SZ_X * sizeof(bf16_t);

  // 1) fp32 -> bf16. Weights transposed; W_Q additionally pre-scaled by
  //    1/sqrt(D_HEAD) * log2(e) so attention scores are exp2-domain ready.
  {
    const int n4 = (int)(SZ_X / 4);
    cvt_f32_bf16<<<(n4 + 255) / 256, 256, 0, stream>>>(x, xb, n4);
    dim3 tgrid(D_MODEL / 32, D_MODEL / 32), tblk(32, 8);
    const float sc2 = 0.125f * 1.4426950408889634f;
    cvt_t_f32_bf16<<<tgrid, tblk, 0, stream>>>(Wq, wqT, D_MODEL, D_MODEL, sc2);
    cvt_t_f32_bf16<<<tgrid, tblk, 0, stream>>>(Wk, wkT, D_MODEL, D_MODEL, 1.f);
    cvt_t_f32_bf16<<<tgrid, tblk, 0, stream>>>(Wv, wvT, D_MODEL, D_MODEL, 1.f);
    cvt_t_f32_bf16<<<tgrid, tblk, 0, stream>>>(Wo, woT, D_MODEL, D_MODEL, 1.f);
  }

  // 2) Q/K/V projections: [4096,1024] @ [1024,1024] -> bf16
  //    V written transposed (Vt[d][s]) for contiguous P@V fragments.
  dim3 gridG(D_MODEL / 64, SEQ / 128);  // (16, 32)
  gemm_bf16_wmma<true, false><<<gridG, 256, 0, stream>>>(
      xb, wqT, Qb, nullptr, SEQ, D_MODEL, D_MODEL);
  gemm_bf16_wmma<true, false><<<gridG, 256, 0, stream>>>(
      xb, wkT, Kb, nullptr, SEQ, D_MODEL, D_MODEL);
  gemm_bf16_wmma<true, true><<<gridG, 256, 0, stream>>>(
      xb, wvT, VbT, nullptr, SEQ, D_MODEL, D_MODEL);

  // 3) causal flash attention: 16 heads * 256 qtiles = 4096 waves / 8 = 512 blocks
  attn_causal_wmma<<<(N_HEADS * (SEQ / 16)) / 8, 256, 0, stream>>>(Qb, Kb, VbT,
                                                                   Ab);

  // 4) output projection -> fp32 d_out
  gemm_bf16_wmma<false, false><<<gridG, 256, 0, stream>>>(
      Ab, woT, nullptr, out, SEQ, D_MODEL, D_MODEL);
}
